// HRMController_19215683682891
// MI455X (gfx1250) — matrix operand order repaired and checked by reading, not compiled, not run on hardware
//
#include <hip/hip_runtime.h>
#include <hip/hip_bf16.h>
#include <stdint.h>

#define D_MODEL 2048
#define D_CTRL  2048
#define BATCH   8192
#define LN_EPS  1e-5f

typedef __attribute__((ext_vector_type(16))) __bf16    v16bf;
typedef __attribute__((ext_vector_type(8)))  float     v8f;
typedef __attribute__((ext_vector_type(4)))  uint32_t  u32x4;
typedef __attribute__((ext_vector_type(8)))  uint32_t  u32x8;

__device__ __forceinline__ unsigned short f2bf(float f) {
    union { float f; uint32_t u; } c; c.f = f;
    uint32_t u = c.u;
    uint32_t r = u + 0x7fffu + ((u >> 16) & 1u);   // round-to-nearest-even
    return (unsigned short)(r >> 16);
}

// ---------------------------------------------------------------------------
// TDM: DMA one [128 rows x 32 k] bf16 tile (row stride = rowlen elements) from
// global into LDS at lds_off, via a 2D Tensor DMA Descriptor.  Issued once per
// wave; tracked with TENSORcnt.
// ---------------------------------------------------------------------------
__device__ __forceinline__ void tdm_load_tile(const unsigned short* gtile,
                                              uint32_t lds_off,
                                              uint32_t rowlen,   // tensor_dim0 & stride (elems)
                                              uint32_t nrows)    // tensor_dim1 (elems)
{
    uint64_t ga = (uint64_t)(uintptr_t)gtile;
    u32x4 g0;
    g0[0] = 1u;                                            // count=1, user mode
    g0[1] = lds_off;                                       // LDS byte address
    g0[2] = (uint32_t)ga;                                  // global addr [31:0]
    g0[3] = ((uint32_t)(ga >> 32) & 0x1FFFFFFu) | (2u << 30); // addr[56:32] | type=2
    u32x8 g1;
    g1[0] = 1u << 16;                                      // data_size=1 (2B), no mask/pad
    g1[1] = (rowlen & 0xFFFFu) << 16;                      // tensor_dim0[15:0]
    g1[2] = (rowlen >> 16) | ((nrows & 0xFFFFu) << 16);    // td0[31:16] | td1[15:0]
    g1[3] = (nrows >> 16) | (32u << 16);                   // td1[31:16] | tile_dim0=32
    g1[4] = 128u;                                          // tile_dim1=128, tile_dim2=0
    g1[5] = rowlen;                                        // tensor_dim0_stride[31:0]
    g1[6] = 0u;                                            // stride[47:32] | dim1_stride lo
    g1[7] = 0u;
    asm volatile("tensor_load_to_lds %0, %1" :: "s"(g0), "s"(g1) : "memory");
}

enum { EP_PLAIN = 0, EP_XPROJ = 1, EP_GATES = 2, EP_CAND = 3 };

// ---------------------------------------------------------------------------
// C = act( [A0|A1] (MxK, bf16) * Bt^T (Bt is [N][K] bf16) + bias )
// Block: 128x128 tile, 256 threads (8 waves 2x4), BK=32, TDM double-buffered.
// ---------------------------------------------------------------------------
template <int EP>
__global__ __launch_bounds__(256)
void gemm_bf16_wmma(const unsigned short* __restrict__ A0, int lda0,
                    const unsigned short* __restrict__ A1, int lda1, int K0,
                    const unsigned short* __restrict__ Bt, int ldb,   // [N][K]
                    const float* __restrict__ bias,
                    int K, int N,
                    void* __restrict__ out0,
                    float* __restrict__ out1,
                    const float* __restrict__ in0,
                    const float* __restrict__ in1)
{
    __shared__ unsigned short As[2][128 * 32];   // [m][k]
    __shared__ unsigned short Bs[2][128 * 32];   // [n][k]

    const int t     = threadIdx.x;
    const int lane  = t & 31;
    const int wid   = t >> 5;
    const int laneM = lane & 15;
    const int laneH = lane >> 4;
    const int bM    = blockIdx.y * 128;
    const int bN    = blockIdx.x * 128;
    const int waveM = (wid >> 2) * 64;
    const int waveN = (wid & 3) * 32;

    const uint32_t AsOff[2] = { (uint32_t)(uintptr_t)&As[0][0], (uint32_t)(uintptr_t)&As[1][0] };
    const uint32_t BsOff[2] = { (uint32_t)(uintptr_t)&Bs[0][0], (uint32_t)(uintptr_t)&Bs[1][0] };

    const v8f vzero = {0.f, 0.f, 0.f, 0.f, 0.f, 0.f, 0.f, 0.f};
    v8f acc[4][2];
#pragma unroll
    for (int i = 0; i < 4; ++i)
#pragma unroll
        for (int j = 0; j < 2; ++j) acc[i][j] = vzero;

    const int nK = K >> 5;
    int cur = 0;

    if (wid == 0) {   // preload K-step 0 into buffer 0
        tdm_load_tile(A0 + (size_t)bM * lda0, AsOff[0], (uint32_t)lda0, BATCH);
        tdm_load_tile(Bt + (size_t)bN * ldb, BsOff[0], (uint32_t)ldb, (uint32_t)N);
    }

    for (int kt = 0; kt < nK; ++kt) {
        if (wid == 0) {
            const int kn = kt + 1;
            if (kn < nK) {           // issue next tile into the other buffer
                const int k0n = kn * 32;
                const unsigned short* Asrc; int lda, kb;
                if (k0n < K0) { Asrc = A0; lda = lda0; kb = k0n; }
                else          { Asrc = A1; lda = lda1; kb = k0n - K0; }
                tdm_load_tile(Asrc + (size_t)bM * lda + kb, AsOff[cur ^ 1], (uint32_t)lda, BATCH);
                tdm_load_tile(Bt + (size_t)bN * ldb + k0n, BsOff[cur ^ 1], (uint32_t)ldb, (uint32_t)N);
                __builtin_amdgcn_s_wait_tensorcnt(2);   // current buffer's 2 DMAs done
            } else {
                __builtin_amdgcn_s_wait_tensorcnt(0);
            }
        }
        __syncthreads();             // current buffer visible to all waves

        union U { uint4 u[2]; v16bf v; };
        U a[4], b[2];
#pragma unroll
        for (int mt = 0; mt < 4; ++mt) {
            const unsigned short* pa = &As[cur][(waveM + mt * 16 + laneM) * 32 + laneH * 8];
            a[mt].u[0] = *(const uint4*)(pa);
            a[mt].u[1] = *(const uint4*)(pa + 16);
        }
#pragma unroll
        for (int nt = 0; nt < 2; ++nt) {
            const unsigned short* pb = &Bs[cur][(waveN + nt * 16 + laneM) * 32 + laneH * 8];
            b[nt].u[0] = *(const uint4*)(pb);
            b[nt].u[1] = *(const uint4*)(pb + 16);
        }
#pragma unroll
        for (int mt = 0; mt < 4; ++mt)
#pragma unroll
            for (int nt = 0; nt < 2; ++nt)
                acc[mt][nt] = __builtin_amdgcn_wmma_f32_16x16x32_bf16(
                    false, a[mt].v, false, b[nt].v, (short)0, acc[mt][nt], false, false);

        __syncthreads();             // everyone done reading before next DMA reuses buffer
        cur ^= 1;
    }

    // Epilogue. C layout: VGPR r -> row (r + laneH*8), col = laneM within 16x16.
    const int HC = N >> 1;
#pragma unroll
    for (int mt = 0; mt < 4; ++mt) {
#pragma unroll
        for (int nt = 0; nt < 2; ++nt) {
            const int col = bN + waveN + nt * 16 + laneM;
            const float bv = bias[col];
#pragma unroll
            for (int r = 0; r < 8; ++r) {
                const int row = bM + waveM + mt * 16 + laneH * 8 + r;
                float v = acc[mt][nt][r] + bv;
                if constexpr (EP == EP_PLAIN) {
                    ((float*)out0)[(size_t)row * N + col] = v;
                } else if constexpr (EP == EP_XPROJ) {
                    ((unsigned short*)out0)[(size_t)row * N + col] = f2bf(v);
                } else if constexpr (EP == EP_GATES) {
                    float s = 1.f / (1.f + __expf(-v));
                    if (col < HC) {   // reset gate -> bf16(r * prev) feeds cand_in
                        ((unsigned short*)out0)[(size_t)row * HC + col] =
                            f2bf(s * in0[(size_t)row * HC + col]);
                    } else {          // update gate (fp32, read by cand epilogue)
                        out1[(size_t)row * HC + (col - HC)] = s;
                    }
                } else {              // EP_CAND: GRU blend -> pre-LN state fp32
                    float c  = tanhf(v);
                    float u  = in0[(size_t)row * N + col];
                    float pv = in1[(size_t)row * N + col];
                    ((float*)out0)[(size_t)row * N + col] = u * pv + (1.f - u) * c;
                }
            }
        }
    }
}

// fp32 -> bf16 bulk convert (4 elems/thread)
__global__ __launch_bounds__(256)
void conv_f32_to_bf16(const float* __restrict__ in, unsigned short* __restrict__ out, int n4)
{
    int i = blockIdx.x * 256 + threadIdx.x;
    if (i < n4) {
        float4 v = *(const float4*)(in + (size_t)i * 4);
        uint32_t lo = (uint32_t)f2bf(v.x) | ((uint32_t)f2bf(v.y) << 16);
        uint32_t hi = (uint32_t)f2bf(v.z) | ((uint32_t)f2bf(v.w) << 16);
        *(uint2*)(out + (size_t)i * 4) = make_uint2(lo, hi);
    }
}

// W[K][N] fp32 -> Wt[N][K] bf16, 32x32 LDS tile transpose
__global__ __launch_bounds__(256)
void transpose_f32_to_bf16(const float* __restrict__ W, unsigned short* __restrict__ Wt,
                           int K, int N)
{
    __shared__ unsigned short T[32][33];
    const int n0 = blockIdx.x * 32, k0 = blockIdx.y * 32;
    const int t = threadIdx.x;
    {
        const int r = t >> 3, cq = (t & 7) * 4;
        float4 v = *(const float4*)(W + (size_t)(k0 + r) * N + n0 + cq);
        T[cq + 0][r] = f2bf(v.x); T[cq + 1][r] = f2bf(v.y);
        T[cq + 2][r] = f2bf(v.z); T[cq + 3][r] = f2bf(v.w);
    }
    __syncthreads();
    {
        const int n = t >> 3, kq = (t & 7) * 4;
        uint32_t lo = (uint32_t)T[n][kq]     | ((uint32_t)T[n][kq + 1] << 16);
        uint32_t hi = (uint32_t)T[n][kq + 2] | ((uint32_t)T[n][kq + 3] << 16);
        *(uint2*)(Wt + (size_t)(n0 + n) * K + k0 + kq) = make_uint2(lo, hi);
    }
}

__global__ __launch_bounds__(256)
void ln_kernel(const float* __restrict__ x, const float* __restrict__ gamma,
               const float* __restrict__ beta, float* __restrict__ y,
               unsigned short* __restrict__ yb, int N)
{
    __shared__ float s_sum[256];
    __shared__ float s_sq[256];
    const int row = blockIdx.x;
    const float* xr = x + (size_t)row * N;
    float sum = 0.f, sq = 0.f;
    for (int c = threadIdx.x; c < N; c += 256) { float v = xr[c]; sum += v; sq += v * v; }
    s_sum[threadIdx.x] = sum; s_sq[threadIdx.x] = sq;
    __syncthreads();
    for (int off = 128; off > 0; off >>= 1) {
        if (threadIdx.x < off) {
            s_sum[threadIdx.x] += s_sum[threadIdx.x + off];
            s_sq[threadIdx.x]  += s_sq[threadIdx.x + off];
        }
        __syncthreads();
    }
    const float mean = s_sum[0] / (float)N;
    const float var  = s_sq[0] / (float)N - mean * mean;
    const float rstd = rsqrtf(var + LN_EPS);
    float* yr = y + (size_t)row * N;
    unsigned short* ybr = yb + (size_t)row * N;
    for (int c = threadIdx.x; c < N; c += 256) {
        float v = (xr[c] - mean) * rstd * gamma[c] + beta[c];
        yr[c]  = v;
        ybr[c] = f2bf(v);
    }
}

extern "C" void kernel_launch(void* const* d_in, const int* in_sizes, int n_in,
                              void* d_out, int out_size, void* d_ws, size_t ws_size,
                              hipStream_t stream)
{
    (void)in_sizes; (void)n_in; (void)out_size; (void)ws_size;

    const float* retnet = (const float*)d_in[0];
    const float* prev   = (const float*)d_in[1];
    const float* W_in   = (const float*)d_in[2];
    const float* b_in   = (const float*)d_in[3];
    const float* W_gate = (const float*)d_in[4];
    const float* b_gate = (const float*)d_in[5];
    const float* W_cand = (const float*)d_in[6];
    const float* b_cand = (const float*)d_in[7];
    const float* W_q    = (const float*)d_in[8];
    const float* b_q    = (const float*)d_in[9];
    const float* W_s    = (const float*)d_in[10];
    const float* b_s    = (const float*)d_in[11];
    const float* gamma  = (const float*)d_in[12];
    const float* beta   = (const float*)d_in[13];

    float* out       = (float*)d_out;
    float* new_state = out;                                 // [B, dc] final
    float* query     = out + (size_t)BATCH * D_MODEL;       // scratch h_pre, then q
    float* outstate  = query + (size_t)BATCH * D_MODEL;     // scratch u, then s

    const size_t BD = (size_t)BATCH * D_CTRL;               // 16M elems
    unsigned short* ws16    = (unsigned short*)d_ws;
    unsigned short* ab_ret  = ws16;                         // [B, dm] bf16
    unsigned short* ab_prev = ab_ret  + BD;                 // [B, dc] bf16
    unsigned short* xb      = ab_prev + BD;                 // [B, dc] bf16
    unsigned short* rpb     = xb      + BD;                 // [B, dc] bf16
    unsigned short* hb      = rpb     + BD;                 // [B, dc] bf16
    unsigned short* wt_in   = hb      + BD;                 // [2048][2048]
    unsigned short* wt_gate = wt_in   + (size_t)D_CTRL * D_MODEL;         // [4096][4096]
    unsigned short* wt_cand = wt_gate + (size_t)2 * D_CTRL * 2 * D_CTRL;  // [2048][4096]
    unsigned short* wt_q    = wt_cand + (size_t)D_CTRL * 2 * D_CTRL;      // [2048][2048]
    unsigned short* wt_s    = wt_q    + (size_t)D_MODEL * D_CTRL;         // [2048][2048]

    const dim3 blk(256);
    const int n4 = (int)(BD / 4);

    // --- one-time fp32 -> bf16 conversions (and weight transposes) ---
    conv_f32_to_bf16<<<dim3((n4 + 255) / 256), blk, 0, stream>>>(retnet, ab_ret, n4);
    conv_f32_to_bf16<<<dim3((n4 + 255) / 256), blk, 0, stream>>>(prev, ab_prev, n4);
    transpose_f32_to_bf16<<<dim3(D_CTRL / 32, D_MODEL / 32), blk, 0, stream>>>(W_in, wt_in, D_MODEL, D_CTRL);
    transpose_f32_to_bf16<<<dim3(2 * D_CTRL / 32, 2 * D_CTRL / 32), blk, 0, stream>>>(W_gate, wt_gate, 2 * D_CTRL, 2 * D_CTRL);
    transpose_f32_to_bf16<<<dim3(D_CTRL / 32, 2 * D_CTRL / 32), blk, 0, stream>>>(W_cand, wt_cand, 2 * D_CTRL, D_CTRL);
    transpose_f32_to_bf16<<<dim3(D_MODEL / 32, D_CTRL / 32), blk, 0, stream>>>(W_q, wt_q, D_CTRL, D_MODEL);
    transpose_f32_to_bf16<<<dim3(D_MODEL / 32, D_CTRL / 32), blk, 0, stream>>>(W_s, wt_s, D_CTRL, D_MODEL);

    // 1) x = retnet @ W_in + b_in    -> bf16 xb
    gemm_bf16_wmma<EP_XPROJ><<<dim3(D_CTRL / 128, BATCH / 128), blk, 0, stream>>>(
        ab_ret, D_MODEL, ab_ret, D_MODEL, D_MODEL, wt_in, D_MODEL, b_in, D_MODEL, D_CTRL,
        xb, nullptr, nullptr, nullptr);

    // 2) gates = [x|prev] @ W_gate + b ; sigmoid -> rpb = bf16(r*prev), u fp32
    gemm_bf16_wmma<EP_GATES><<<dim3(2 * D_CTRL / 128, BATCH / 128), blk, 0, stream>>>(
        xb, D_CTRL, ab_prev, D_CTRL, D_CTRL, wt_gate, 2 * D_CTRL, b_gate, 2 * D_CTRL, 2 * D_CTRL,
        rpb, outstate /*u*/, prev, nullptr);

    // 3) cand = tanh([x|rp] @ W_cand + b); GRU blend -> pre-LN fp32 in `query`
    gemm_bf16_wmma<EP_CAND><<<dim3(D_CTRL / 128, BATCH / 128), blk, 0, stream>>>(
        xb, D_CTRL, rpb, D_CTRL, D_CTRL, wt_cand, 2 * D_CTRL, b_cand, 2 * D_CTRL, D_CTRL,
        query /*h_pre*/, nullptr, outstate /*u*/, prev);

    // 4) LayerNorm -> new_state fp32 (out slot 0) + hb bf16
    ln_kernel<<<BATCH, 256, 0, stream>>>(query, gamma, beta, new_state, hb, D_CTRL);

    // 5) query = h @ W_q + b_q
    gemm_bf16_wmma<EP_PLAIN><<<dim3(D_MODEL / 128, BATCH / 128), blk, 0, stream>>>(
        hb, D_CTRL, hb, D_CTRL, D_CTRL, wt_q, D_CTRL, b_q, D_CTRL, D_MODEL,
        query, nullptr, nullptr, nullptr);

    // 6) output_state = h @ W_s + b_s
    gemm_bf16_wmma<EP_PLAIN><<<dim3(D_MODEL / 128, BATCH / 128), blk, 0, stream>>>(
        hb, D_CTRL, hb, D_CTRL, D_CTRL, wt_s, D_CTRL, b_s, D_CTRL, D_MODEL,
        outstate, nullptr, nullptr, nullptr);
}
